// SemanticActor_11948599017715
// MI455X (gfx1250) — matrix operand order repaired and checked by reading, not compiled, not run on hardware
//
#include <hip/hip_runtime.h>

typedef _Float16 v16h __attribute__((ext_vector_type(16)));
typedef _Float16 v8h  __attribute__((ext_vector_type(8)));
typedef __fp16   h2   __attribute__((ext_vector_type(2)));   // cvt_pkrtz result type
typedef float    v8f  __attribute__((ext_vector_type(8)));

#define BATCH    65536
#define ROWS     128      // batch rows per workgroup
#define THREADS  512      // 16 waves (wave32)
#define NPAIRS   6

// LDS strides in halves (padded +16 halves => 32B-aligned rows, bank spread)
#define SW2 272   // W2T / WrT rows (K up to 256)
#define SW1 112   // W1T rows (K padded to 96; row 80 = bias)
#define SWH 272   // head-weight rows (K = 256)
#define SH  272   // activation rows (256 cols)
#define SI  112   // pair-input rows (96 cols; col 80 = 1.0)

// LDS byte layout
#define OFF_W2  0
#define SZ_W2   (256*SW2*2)            // 139264
#define OFF_W1  (OFF_W2 + SZ_W2)
#define SZ_W1   (256*SW1*2)            // 57344 (reused for head weights later)
#define OFF_H1  (OFF_W1 + SZ_W1)
#define SZ_H1   (ROWS*SH*2)            // 69632 (h1 / agg / hr, sequentially)
#define OFF_INP (OFF_H1 + SZ_H1)
#define SZ_INP  (ROWS*SI*2)            // 28672
#define OFF_B   (OFF_INP + SZ_INP)
#define SZ_B    ((2*256 + 16)*4)       // b2, br + head bias(16)  (b1 folded into W1T)
#define SMEM_TOTAL (OFF_B + SZ_B)      // ~297KB < 320KB WGP LDS

__constant__ int cFIRST[20]  = {0,1,2,3,4,5,6,7,8,9,10,11,12,13,15,16,17,20,21,25};
__constant__ int cSECOND[20] = {0,1,2,3,4,5,6,7,8,9,14,18,19,22,23,24,26,27,28,29};
__constant__ int cPI[6] = {0,0,1,1,2,2};
__constant__ int cPJ[6] = {1,2,0,2,0,1};
__constant__ int cPS[6] = {0,0,1,0,1,1};   // 0 = FIRST inds, 1 = SECOND inds

static __device__ __forceinline__ v16h mkfrag(v8h lo, v8h hi) {
    v16h r;
#pragma unroll
    for (int i = 0; i < 8; ++i) { r[i] = lo[i]; r[i + 8] = hi[i]; }
    return r;
}

// A-fragment (16x32 f16): rowp already includes (lane>=16 ? 8 : 0) K-offset.
// halves 0..7  = K = ks*32 + lh*8 + {0..7}
// halves 8..15 = K = ks*32 + lh*8 + 16 + {0..7}
static __device__ __forceinline__ v16h load_a(const _Float16* rowp, int ks) {
    v8h lo = *(const v8h*)(rowp + ks * 32);
    v8h hi = *(const v8h*)(rowp + ks * 32 + 16);
    return mkfrag(lo, hi);
}

// B-fragment (32x16 f16) from transposed weights WT[n][k]:
// lane n = nT + (lane&15); halves 0..7 = K = ks*32 + lh*8 + {0..7}, halves 8..15 = K = ks*32 + lh*8 + 16 + {0..7}
static __device__ __forceinline__ v16h load_b(const _Float16* wt, int stride,
                                              int nT, int ks, int ll, int lh) {
    const _Float16* p = wt + (nT + ll) * stride + ks * 32 + lh * 8;
    return mkfrag(*(const v8h*)p, *(const v8h*)(p + 16));
}

static __device__ __forceinline__ v8f wmma_f16(v16h a, v16h b, v8f c) {
    return __builtin_amdgcn_wmma_f32_16x16x32_f16(false, a, false, b, (short)0, c, false, false);
}

// single-instruction ReLU: v_med3_num_f32 (no canonicalize pre-op)
static __device__ __forceinline__ float relu1(float x) {
    return __builtin_amdgcn_fmed3f(x, 0.f, 3.0e38f);
}

// convert a fragment pair (rows r, r+1) and store to row-major f16 LDS
static __device__ __forceinline__ void store2(_Float16* p, float a, float b) {
    h2 pk = __builtin_amdgcn_cvt_pkrtz(a, b);   // v_cvt_pk_rtz_f16_f32
    p[0]  = (_Float16)pk[0];
    p[SH] = (_Float16)pk[1];
}

__global__ __launch_bounds__(THREADS, 1)
void semantic_actor_kernel(const float* __restrict__ obs, const float* __restrict__ ag,
                           const float* __restrict__ g,
                           const float* __restrict__ w1, const float* __restrict__ b1,
                           const float* __restrict__ w2, const float* __restrict__ b2,
                           const float* __restrict__ rw, const float* __restrict__ rb,
                           const float* __restrict__ mw, const float* __restrict__ mb,
                           const float* __restrict__ lw, const float* __restrict__ lb,
                           float* __restrict__ out) {
    extern __shared__ char smem[];
    _Float16* W2T = (_Float16*)(smem + OFF_W2);   // later: rho weights (transposed)
    _Float16* W1T = (_Float16*)(smem + OFF_W1);   // later: head weights (transposed)
    _Float16* H1  = (_Float16*)(smem + OFF_H1);   // h1 -> agg -> hr (f16, row-major)
    _Float16* INP = (_Float16*)(smem + OFF_INP);  // pair input (f16, K padded to 96)
    float* B2 = (float*)(smem + OFF_B);
    float* BR = B2 + 256;
    float* BH = BR + 256;                          // 16 head biases
    _Float16* WHT = W1T;                           // alias, stride SWH

    const int tid  = threadIdx.x;
    const int lane = tid & 31;
    const int ll   = lane & 15;     // 0..15  (N column / A row within tile)
    const int lh   = lane >> 4;     // 0/1    (half-wave select)
    const int wave = tid >> 5;      // 0..15
    const int wrow = wave >> 1;     // 0..7   row-tile slice
    const int wcol = wave & 1;      // 0..1   column half
    const int m0   = wrow * 16;
    const int n0   = wcol * 128;
    const int rowBase = blockIdx.x * ROWS;

    // ---------------- stage phi weights + biases ----------------
    for (int idx = tid; idx < 80 * 256; idx += THREADS) {       // coalesced read: n fastest
        int k = idx >> 8, n = idx & 255;
        W1T[n * SW1 + k] = (_Float16)w1[idx];
    }
    for (int idx = tid; idx < 16 * 256; idx += THREADS) {       // K rows 80..95
        int k = 80 + (idx >> 8), n = idx & 255;
        W1T[n * SW1 + k] = (k == 80) ? (_Float16)b1[n] : (_Float16)0.f;  // bias folded as K=80
    }
    for (int idx = tid; idx < 256 * 256; idx += THREADS) {
        int k = idx >> 8, n = idx & 255;
        W2T[n * SW2 + k] = (_Float16)w2[idx];
    }
    for (int i = tid; i < 256; i += THREADS) { B2[i] = b2[i]; BR[i] = rb[i]; }
    if (tid < 16) BH[tid] = (tid < 4) ? mb[tid] : (tid < 8 ? lb[tid - 4] : 0.f);

    // persistent pair-sum accumulators: 8 col-tiles x 16x16 f32 fragment
    v8f agg[8];
#pragma unroll
    for (int t = 0; t < 8; ++t)
#pragma unroll
        for (int e = 0; e < 8; ++e) agg[t][e] = 0.f;

    // ---------------- pair loop ----------------
    for (int p = 0; p < NPAIRS; ++p) {
        const int oi = cPI[p], oj = cPJ[p];
        const int* IDX = cPS[p] ? cSECOND : cFIRST;

        // stage pair input (safe: previous pair's layer1 finished before the
        // barrier that preceded its layer2, and layer2 touches only H1/W2T)
        for (int idx = tid; idx < ROWS * 96; idx += THREADS) {
            int r = idx / 96, c = idx - r * 96;
            int gr = rowBase + r;
            float v;
            if      (c < 20) v = ag[gr * 30 + IDX[c]];
            else if (c < 30) v = obs[gr * 55 + (c - 20)];
            else if (c < 50) v = g[gr * 30 + IDX[c - 30]];
            else if (c < 65) v = obs[gr * 55 + 10 + 15 * oi + (c - 50)];
            else if (c < 80) v = obs[gr * 55 + 10 + 15 * oj + (c - 65)];
            else if (c == 80) v = 1.f;                 // bias lane (pairs W1T row 80)
            else             v = 0.f;
            INP[r * SI + c] = (_Float16)v;
        }
        __syncthreads();

        // ---- layer 1: h1 = relu(inp @ [W1;b1]), K=96 (3 WMMA steps, C starts 0) ----
        v16h a1[3];
        {
            const _Float16* ap = INP + (m0 + ll) * SI + lh * 8;
#pragma unroll
            for (int ks = 0; ks < 3; ++ks) a1[ks] = load_a(ap, ks);
        }
#pragma unroll
        for (int ct = 0; ct < 8; ++ct) {
            const int nT = n0 + ct * 16;
            v8f c;
#pragma unroll
            for (int e = 0; e < 8; ++e) c[e] = 0.f;     // folds to inline-0 SRC2
#pragma unroll
            for (int ks = 0; ks < 3; ++ks)
                c = wmma_f16(a1[ks], load_b(W1T, SW1, nT, ks, ll, lh), c);
            _Float16* hp = H1 + (m0 + lh * 8) * SH + nT + ll;
#pragma unroll
            for (int r = 0; r < 8; r += 2)
                store2(hp + r * SH, relu1(c[r]), relu1(c[r + 1]));
        }
        __syncthreads();

        // ---- layer 2: agg += relu(h1 @ W2 + b2), K=256 (8 WMMA steps) ----
        v16h a2[8];
        {
            const _Float16* ap = H1 + (m0 + ll) * SH + lh * 8;
#pragma unroll
            for (int ks = 0; ks < 8; ++ks) a2[ks] = load_a(ap, ks);
        }
#pragma unroll
        for (int ct = 0; ct < 8; ++ct) {
            const int nT = n0 + ct * 16;
            float bias = B2[nT + ll];
            v8f c;
#pragma unroll
            for (int e = 0; e < 8; ++e) c[e] = bias;
#pragma unroll
            for (int ks = 0; ks < 8; ++ks)
                c = wmma_f16(a2[ks], load_b(W2T, SW2, nT, ks, ll, lh), c);
#pragma unroll
            for (int r = 0; r < 8; ++r) agg[ct][r] += relu1(c[r]);
        }
    }

    // ---------------- rho layer setup ----------------
    __syncthreads();  // all waves done with H1 (layer2 reads) and W2T
    {   // store agg (f16) into H1 region; stage rho + head weights
#pragma unroll
        for (int ct = 0; ct < 8; ++ct) {
            _Float16* hp = H1 + (m0 + lh * 8) * SH + n0 + ct * 16 + ll;
#pragma unroll
            for (int r = 0; r < 8; r += 2)
                store2(hp + r * SH, agg[ct][r], agg[ct][r + 1]);
        }
    }
    for (int idx = tid; idx < 256 * 256; idx += THREADS) {
        int k = idx >> 8, n = idx & 255;
        W2T[n * SW2 + k] = (_Float16)rw[idx];
    }
    for (int idx = tid; idx < 16 * 256; idx += THREADS) {
        int n = idx >> 8, k = idx & 255;
        float v = (n < 4) ? mw[k * 4 + n] : (n < 8 ? lw[k * 4 + (n - 4)] : 0.f);
        WHT[n * SWH + k] = (_Float16)v;
    }
    __syncthreads();

    // ---- rho: hr = relu(agg @ Wr + br), K=256 ----
    v16h a3[8];
    {
        const _Float16* ap = H1 + (m0 + ll) * SH + lh * 8;
#pragma unroll
        for (int ks = 0; ks < 8; ++ks) a3[ks] = load_a(ap, ks);
    }
    v8f c3[8];
#pragma unroll
    for (int ct = 0; ct < 8; ++ct) {
        const int nT = n0 + ct * 16;
        float bias = BR[nT + ll];
        v8f c;
#pragma unroll
        for (int e = 0; e < 8; ++e) c[e] = bias;
#pragma unroll
        for (int ks = 0; ks < 8; ++ks)
            c = wmma_f16(a3[ks], load_b(W2T, SW2, nT, ks, ll, lh), c);
#pragma unroll
        for (int r = 0; r < 8; ++r) c3[ct][r] = relu1(c[r]);
    }
    __syncthreads();  // all rho A-fragment reads of agg complete
    {
#pragma unroll
        for (int ct = 0; ct < 8; ++ct) {
            _Float16* hp = H1 + (m0 + lh * 8) * SH + n0 + ct * 16 + ll;
#pragma unroll
            for (int r = 0; r < 8; r += 2)
                store2(hp + r * SH, c3[ct][r], c3[ct][r + 1]);
        }
    }
    __syncthreads();

    // ---- heads: [mean | logstd | 0...] = hr @ Wh (256x16), one col-tile ----
    if (wcol == 0) {   // wave-uniform predicate: EXEC stays all-ones for WMMA
        v16h a4[8];
        const _Float16* ap = H1 + (m0 + ll) * SH + lh * 8;
#pragma unroll
        for (int ks = 0; ks < 8; ++ks) a4[ks] = load_a(ap, ks);
        float bias = BH[ll];
        v8f c;
#pragma unroll
        for (int e = 0; e < 8; ++e) c[e] = bias;
#pragma unroll
        for (int ks = 0; ks < 8; ++ks)
            c = wmma_f16(a4[ks], load_b(WHT, SWH, 0, ks, ll, lh), c);
        const int n = ll;
#pragma unroll
        for (int r = 0; r < 8; ++r) {
            int gr = rowBase + m0 + lh * 8 + r;
            float v = c[r];
            if (n < 4) {
                out[gr * 4 + n] = v;                               // mean
            } else if (n < 8) {
                out[BATCH * 4 + gr * 4 + (n - 4)] =
                    __builtin_amdgcn_fmed3f(v, -20.f, 2.f);        // one-op clip
            }
        }
    }
}

extern "C" void kernel_launch(void* const* d_in, const int* in_sizes, int n_in,
                              void* d_out, int out_size, void* d_ws, size_t ws_size,
                              hipStream_t stream) {
    (void)in_sizes; (void)n_in; (void)out_size; (void)d_ws; (void)ws_size;
    const float* obs = (const float*)d_in[0];
    const float* ag  = (const float*)d_in[1];
    const float* g   = (const float*)d_in[2];
    const float* w1  = (const float*)d_in[3];
    const float* b1  = (const float*)d_in[4];
    const float* w2  = (const float*)d_in[5];
    const float* b2  = (const float*)d_in[6];
    const float* rw  = (const float*)d_in[7];
    const float* rb  = (const float*)d_in[8];
    const float* mw  = (const float*)d_in[9];
    const float* mb  = (const float*)d_in[10];
    const float* lw  = (const float*)d_in[11];
    const float* lb  = (const float*)d_in[12];
    float* out = (float*)d_out;

    (void)hipFuncSetAttribute((const void*)semantic_actor_kernel,
                              hipFuncAttributeMaxDynamicSharedMemorySize, SMEM_TOTAL);
    dim3 grid(BATCH / ROWS);   // 512 workgroups
    dim3 block(THREADS);       // 16 wave32 waves
    semantic_actor_kernel<<<grid, block, SMEM_TOTAL, stream>>>(
        obs, ag, g, w1, b1, w2, b2, rw, rb, mw, mb, lw, lb, out);
}
